// FB_FMM_41102837023336
// MI455X (gfx1250) — matrix operand order repaired and checked by reading, not compiled, not run on hardware
//
#include <hip/hip_runtime.h>
#include <math.h>

// ---------------------------------------------------------------------------
// MI455X (gfx1250) implementation.
//  - Flash-attention with v_wmma_f32_16x16x32_f16 (wave32), f32 accum,
//    f32 online softmax. Scores never touch HBM.
//  - Q/K row-major (N x 32) f16, V transposed (C x N) f16 so every WMMA
//    A/B fragment is a contiguous 16B/32B load per lane (k = 16h + s in
//    both A and B, not the ISA table order).
//  - P (probabilities) transposed to A-fragment layout through per-wave
//    private LDS; in-wave ordering via s_wait_dscnt 0.
// ---------------------------------------------------------------------------

typedef __attribute__((ext_vector_type(16))) _Float16 v16h;
typedef __attribute__((ext_vector_type(8)))  _Float16 v8h;
typedef __attribute__((ext_vector_type(8)))  float    v8f;

union ABf16 { v16h v; v8h h[2]; };

static __device__ __forceinline__ v8f wmma_f16(v16h a, v16h b, v8f c) {
  return __builtin_amdgcn_wmma_f32_16x16x32_f16(false, a, false, b,
                                                (short)0, c, false, false);
}

#define N_  4096
#define C_  256
#define D_  32
#define B_  2
#define BM_ 64
#define BN_ 64

// ---------------------------------------------------------------------------
// Q/K projection: out[n][d] = f16( maskf(n) * sum_c w[d][c]*x[c][n] + b[d] )
// block 256 = 8 n x 32 d ; grid (N/8, B)
// ---------------------------------------------------------------------------
__global__ __launch_bounds__(256) void proj_qk_kernel(
    const float* __restrict__ x, const float* __restrict__ w,
    const float* __restrict__ bias, const float* __restrict__ mask,
    int maskmode, _Float16* __restrict__ out) {
  int d  = threadIdx.x & 31;
  int nl = threadIdx.x >> 5;
  int n  = blockIdx.x * 8 + nl;
  int b  = blockIdx.y;
  const float* xp = x + (size_t)b * C_ * N_ + n;
  const float* wp = w + (size_t)d * C_;
  float acc = 0.f;
#pragma unroll 8
  for (int c = 0; c < C_; ++c) acc = fmaf(wp[c], xp[(size_t)c * N_], acc);
  float f = 1.f;
  if (maskmode == 1) f = mask[(size_t)b * N_ + n];
  else if (maskmode == 2) f = 1.f - mask[(size_t)b * N_ + n];
  out[((size_t)b * N_ + n) * D_ + d] = (_Float16)(f * acc + bias[d]);
}

// ---------------------------------------------------------------------------
// V projection (transposed output): VT[c][n] = f16( maskf(n)*dot + b[c] )
// block 256 n ; grid (N/256, C, B)
// ---------------------------------------------------------------------------
__global__ __launch_bounds__(256) void proj_v_kernel(
    const float* __restrict__ x, const float* __restrict__ w,
    const float* __restrict__ bias, const float* __restrict__ mask,
    int maskmode, _Float16* __restrict__ vt) {
  int n = blockIdx.x * 256 + threadIdx.x;
  int c = blockIdx.y;
  int b = blockIdx.z;
  const float* xp = x + (size_t)b * C_ * N_ + n;
  const float* wp = w + (size_t)c * C_;
  float acc = 0.f;
#pragma unroll 8
  for (int cc = 0; cc < C_; ++cc) acc = fmaf(wp[cc], xp[(size_t)cc * N_], acc);
  float f = 1.f;
  if (maskmode == 1) f = mask[(size_t)b * N_ + n];
  else if (maskmode == 2) f = 1.f - mask[(size_t)b * N_ + n];
  vt[((size_t)b * C_ + c) * N_ + n] = (_Float16)(f * acc + bias[c]);
}

// ---------------------------------------------------------------------------
// Flash attention: O[b][n][c] = sum_j softmax_j(q_n . k_j) * V[j][c]
// grid (N/BM, B) x 256 threads (8 wave32). Wave w: M-tile (w&3), C-half (w>>2).
// ---------------------------------------------------------------------------
__global__ __launch_bounds__(256) void attn_kernel(
    const _Float16* __restrict__ Q, const _Float16* __restrict__ K,
    const _Float16* __restrict__ VT, float* __restrict__ O) {
  const int b    = blockIdx.y;
  const int lane = threadIdx.x & 31;
  const int w    = threadIdx.x >> 5;
  const int mw   = w & 3;
  const int ch   = w >> 2;
  const int hi   = lane >> 4;   // half-wave select
  const int ln   = lane & 15;
  const int q0   = blockIdx.x * BM_ + mw * 16;
  const int c0   = ch * 128;

  __shared__ _Float16 Plds[8 * 16 * 64];   // 16 KB, per-wave private 16x64
  _Float16* myP = &Plds[w * 16 * 64];

  // A-fragment for Q (16x32 f16): lane lo -> k{0..15}, hi -> k{16..31}
  ABf16 qf;
  {
    const _Float16* qp = Q + ((size_t)b * N_ + (q0 + ln)) * D_;
    qf.h[0] = *(const v8h*)(qp + hi * 16);
    qf.h[1] = *(const v8h*)(qp + 8 + hi * 16);
  }

  const v8f zf = {0.f, 0.f, 0.f, 0.f, 0.f, 0.f, 0.f, 0.f};
  v8f o[8];
#pragma unroll
  for (int t = 0; t < 8; ++t) o[t] = zf;
  float rmax[8], rsum[8];
#pragma unroll
  for (int r = 0; r < 8; ++r) { rmax[r] = -3.0e38f; rsum[r] = 0.f; }

  for (int kb = 0; kb < N_; kb += BN_) {
    if (kb + BN_ < N_)
      __builtin_prefetch(K + ((size_t)b * N_ + (kb + BN_ + lane)) * D_, 0, 1);

    // ---- S = Q (16x32) x K^T (32x64): 4 WMMA ----
    v8f S[4];
#pragma unroll
    for (int t = 0; t < 4; ++t) {
      // B-fragment of K^T straight from row-major K: one 32B load per lane
      const _Float16* kp = K + ((size_t)b * N_ + (kb + t * 16 + ln)) * D_;
      ABf16 kf; kf.v = *(const v16h*)(kp + hi * 16);
      S[t] = wmma_f16(qf.v, kf.v, zf);
    }

    // ---- online softmax over this 64-key block ----
    float alpha[8];
#pragma unroll
    for (int r = 0; r < 8; ++r) {
      float mb = fmaxf(fmaxf(S[0][r], S[1][r]), fmaxf(S[2][r], S[3][r]));
#pragma unroll
      for (int m = 1; m < 16; m <<= 1) mb = fmaxf(mb, __shfl_xor(mb, m));
      float mn = fmaxf(rmax[r], mb);
      alpha[r] = __expf(rmax[r] - mn);
      rmax[r]  = mn;
      float s = 0.f;
#pragma unroll
      for (int t = 0; t < 4; ++t) {
        float p = __expf(S[t][r] - mn);
        S[t][r] = p;
        s += p;
      }
#pragma unroll
      for (int m = 1; m < 16; m <<= 1) s += __shfl_xor(s, m);
      rsum[r] = rsum[r] * alpha[r] + s;
    }
#pragma unroll
    for (int t = 0; t < 8; ++t)
#pragma unroll
      for (int r = 0; r < 8; ++r) o[t][r] *= alpha[r];

    // ---- transpose P to A-fragment layout via private LDS ----
#pragma unroll
    for (int t = 0; t < 4; ++t)
#pragma unroll
      for (int r = 0; r < 8; ++r)
        myP[(r + 8 * hi) * 64 + t * 16 + ln] = (_Float16)S[t][r];
    asm volatile("s_wait_dscnt 0" ::: "memory");

    // ---- O += P (16x64) x V (64x128): 2 K-chunks x 8 col tiles = 16 WMMA ----
#pragma unroll
    for (int kc = 0; kc < 2; ++kc) {
      ABf16 pf;
      const _Float16* pp = myP + ln * 64 + kc * 32 + hi * 16;
      pf.h[0] = *(const v8h*)(pp);
      pf.h[1] = *(const v8h*)(pp + 8);
#pragma unroll
      for (int t = 0; t < 8; ++t) {
        const _Float16* vp = VT + ((size_t)b * C_ + (c0 + t * 16 + ln)) * N_
                                + kb + kc * 32 + hi * 16;
        ABf16 vf; vf.v = *(const v16h*)vp;
        o[t] = wmma_f16(pf.v, vf.v, o[t]);
      }
    }
  }

  // ---- epilogue: divide by softmax denominator, write (B,N,C) ----
  float inv[8];
#pragma unroll
  for (int r = 0; r < 8; ++r) inv[r] = 1.f / rsum[r];
#pragma unroll
  for (int t = 0; t < 8; ++t)
#pragma unroll
    for (int r = 0; r < 8; ++r)
      O[((size_t)b * N_ + q0 + r + 8 * hi) * C_ + c0 + t * 16 + ln] =
          o[t][r] * inv[r];
}

// ---------------------------------------------------------------------------
// x1[b,c,n] = sa_gamma * O1[b,n,c] + x[b,c,n]
// ---------------------------------------------------------------------------
__global__ __launch_bounds__(256) void pw1_kernel(
    const float* __restrict__ x, const float* __restrict__ O1,
    const float* __restrict__ sag, float* __restrict__ x1) {
  size_t idx = (size_t)blockIdx.x * 256 + threadIdx.x;
  int n = (int)(idx & (N_ - 1));
  int c = (int)((idx >> 12) & (C_ - 1));
  int b = (int)(idx >> 20);
  x1[idx] = sag[0] * O1[((size_t)b * N_ + n) * C_ + c] + x[idx];
}

// ---------------------------------------------------------------------------
// per (b,c): R = sqrt(var(sw_bg)+eps) / sqrt(var(feature_f)+eps), ddof=1
// feature_f = mask * x1 ; sw_bg = O2 (stored B,N,C)
// ---------------------------------------------------------------------------
__global__ __launch_bounds__(256) void std_kernel(
    const float* __restrict__ x1, const float* __restrict__ mask,
    const float* __restrict__ O2, float* __restrict__ R) {
  int c = blockIdx.x, b = blockIdx.y, tid = threadIdx.x;
  float sf = 0.f, sff = 0.f, sg = 0.f, sgg = 0.f;
  for (int n = tid; n < N_; n += 256) {
    float f = mask[(size_t)b * N_ + n] * x1[((size_t)b * C_ + c) * N_ + n];
    sf += f; sff += f * f;
    float g = O2[((size_t)b * N_ + n) * C_ + c];
    sg += g; sgg += g * g;
  }
  __shared__ float red[4][256];
  red[0][tid] = sf; red[1][tid] = sff; red[2][tid] = sg; red[3][tid] = sgg;
  __syncthreads();
  for (int s = 128; s > 0; s >>= 1) {
    if (tid < s) {
      red[0][tid] += red[0][tid + s];
      red[1][tid] += red[1][tid + s];
      red[2][tid] += red[2][tid + s];
      red[3][tid] += red[3][tid + s];
    }
    __syncthreads();
  }
  if (tid == 0) {
    const float n = (float)N_;
    float varf = (red[1][0] - red[0][0] * red[0][0] / n) / (n - 1.f);
    float varg = (red[3][0] - red[2][0] * red[2][0] / n) / (n - 1.f);
    R[(size_t)b * C_ + c] = sqrtf(varg + 1e-5f) / sqrtf(varf + 1e-5f);
  }
}

// ---------------------------------------------------------------------------
// out = x1 + gamma * (mask*x1) * R[b,c]
// ---------------------------------------------------------------------------
__global__ __launch_bounds__(256) void final_kernel(
    const float* __restrict__ x1, const float* __restrict__ mask,
    const float* __restrict__ R, const float* __restrict__ gam,
    float* __restrict__ out) {
  size_t idx = (size_t)blockIdx.x * 256 + threadIdx.x;
  int n = (int)(idx & (N_ - 1));
  int c = (int)((idx >> 12) & (C_ - 1));
  int b = (int)(idx >> 20);
  float xv = x1[idx];
  float f  = mask[(size_t)b * N_ + n] * xv;
  out[idx] = xv + gam[0] * f * R[(size_t)b * C_ + c];
}

// ---------------------------------------------------------------------------
extern "C" void kernel_launch(void* const* d_in, const int* in_sizes, int n_in,
                              void* d_out, int out_size, void* d_ws, size_t ws_size,
                              hipStream_t stream) {
  (void)in_sizes; (void)n_in; (void)out_size; (void)ws_size;
  const float* x     = (const float*)d_in[0];
  const float* mask  = (const float*)d_in[1];
  const float* sa_wq = (const float*)d_in[2];
  const float* sa_bq = (const float*)d_in[3];
  const float* sa_wk = (const float*)d_in[4];
  const float* sa_bk = (const float*)d_in[5];
  const float* sa_wv = (const float*)d_in[6];
  const float* sa_bv = (const float*)d_in[7];
  const float* sa_g  = (const float*)d_in[8];
  const float* wq    = (const float*)d_in[9];
  const float* bq    = (const float*)d_in[10];
  const float* wk    = (const float*)d_in[11];
  const float* bk    = (const float*)d_in[12];
  const float* wv    = (const float*)d_in[13];
  const float* bv    = (const float*)d_in[14];
  const float* gam   = (const float*)d_in[15];

  char* ws = (char*)d_ws;
  const size_t szQK = (size_t)B_ * N_ * D_ * sizeof(_Float16);   // 512 KB
  const size_t szVT = (size_t)B_ * C_ * N_ * sizeof(_Float16);   // 4 MB
  const size_t szO  = (size_t)B_ * N_ * C_ * sizeof(float);      // 8 MB
  _Float16* Q16 = (_Float16*)(ws);
  _Float16* K16 = (_Float16*)(ws + szQK);
  _Float16* VT16 = (_Float16*)(ws + 2 * szQK);
  float* O1 = (float*)(ws + 2 * szQK + szVT);
  float* x1 = (float*)(ws + 2 * szQK + szVT + szO);
  float* O2 = (float*)(ws + 2 * szQK + szVT + 2 * szO);
  float* R  = (float*)(ws + 2 * szQK + szVT + 3 * szO);

  dim3 gQK(N_ / 8, B_), gV(N_ / 256, C_, B_), gA(N_ / BM_, B_), gS(C_, B_);
  int gPW = (B_ * C_ * N_) / 256;

  // ---- self attention ----
  proj_qk_kernel<<<gQK, 256, 0, stream>>>(x, sa_wq, sa_bq, mask, 0, Q16);
  proj_qk_kernel<<<gQK, 256, 0, stream>>>(x, sa_wk, sa_bk, mask, 0, K16);
  proj_v_kernel <<<gV, 256, 0, stream>>>(x, sa_wv, sa_bv, mask, 0, VT16);
  attn_kernel   <<<gA, 256, 0, stream>>>(Q16, K16, VT16, O1);
  pw1_kernel    <<<gPW, 256, 0, stream>>>(x, O1, sa_g, x1);

  // ---- masked cross attention (q: mask*x1 ; k,v: (1-mask)*x1) ----
  proj_qk_kernel<<<gQK, 256, 0, stream>>>(x1, wq, bq, mask, 1, Q16);
  proj_qk_kernel<<<gQK, 256, 0, stream>>>(x1, wk, bk, mask, 2, K16);
  proj_v_kernel <<<gV, 256, 0, stream>>>(x1, wv, bv, mask, 2, VT16);
  attn_kernel   <<<gA, 256, 0, stream>>>(Q16, K16, VT16, O2);

  // ---- FMM rescale + residual ----
  std_kernel  <<<gS, 256, 0, stream>>>(x1, mask, O2, R);
  final_kernel<<<gPW, 256, 0, stream>>>(x1, mask, R, gam, (float*)d_out);
}
